// GNN_89687507076125
// MI455X (gfx1250) — matrix operand order repaired and checked by reading, not compiled, not run on hardware
//
#include <hip/hip_runtime.h>
#include <hip/hip_bf16.h>

typedef __attribute__((ext_vector_type(16))) _Float16 v16h;
typedef __attribute__((ext_vector_type(8)))  _Float16 v8h;
typedef __attribute__((ext_vector_type(8)))  float    v8f;

#define N_NODES  50000
#define N_EDGES  800000
#define IN_DIM   128
#define HID      256
#define N_GRAPHS 512
#define BN_EPS   1e-5f

// ---------------- degree / dinv ----------------
__global__ void k_init_deg(float* deg, int n) {
    int i = blockIdx.x * blockDim.x + threadIdx.x;
    if (i < n) deg[i] = 1.0f;   // self-loop
}
__global__ void k_deg_accum(const int* __restrict__ tgt, float* deg, int e) {
    int i = blockIdx.x * blockDim.x + threadIdx.x;
    if (i < e) atomicAdd(&deg[tgt[i]], 1.0f);
}
__global__ void k_deg_to_dinv(float* deg, int n) {
    int i = blockIdx.x * blockDim.x + threadIdx.x;
    if (i < n) deg[i] = rsqrtf(deg[i]);
}

// ---------------- f16 staging ----------------
__global__ void k_f32_to_f16(const float* __restrict__ in, _Float16* __restrict__ out, int n) {
    int i = blockIdx.x * blockDim.x + threadIdx.x;
    if (i < n) out[i] = (_Float16)in[i];
}
// W is [K,N] row-major f32; produce Wt [N,K] f16 so B fragments are K-contiguous per lane
__global__ void k_transpose_f16(const float* __restrict__ W, _Float16* __restrict__ Wt, int K, int N) {
    int i = blockIdx.x * blockDim.x + threadIdx.x;
    if (i >= K * N) return;
    int k = i / N, n = i % N;
    Wt[n * K + k] = (_Float16)W[i];
}

// ---------------- WMMA GEMM: C[M,N] = A[M,K] (f16) * Bt[N,K]^T (f16), f32 accum ----------------
// block = 512 threads = 16 waves; wave w computes the 16x16 tile (blockIdx.x, w). N must be 256.
__global__ void k_wmma_gemm(const _Float16* __restrict__ A, const _Float16* __restrict__ Bt,
                            float* __restrict__ C, int M, int N, int K) {
    const int wave = threadIdx.x >> 5;
    const int lane = threadIdx.x & 31;
    const int mTile = blockIdx.x;
    const int nTile = wave;                     // 16 waves == 16 N tiles (N==256)
    const int row  = mTile * 16 + (lane & 15);  // A fragment: lane -> matrix row
    const int col  = nTile * 16 + (lane & 15);  // B fragment: lane -> matrix column
    const int kbA  = (lane >> 4) * 16;          // A: lanes 0-15 -> K 0..15; 16-31 -> K 16..31
    const int kbB  = (lane >> 4) * 16;          // B: lanes 0-15 -> K 0..15; 16-31 -> K 16..31

    v8f acc = {};
    for (int k0 = 0; k0 < K; k0 += 32) {
        const _Float16* pa = A + (size_t)row * K + k0 + kbA;
        v16h a = *(const v16h*)(pa);
        v16h b = *(const v16h*)(Bt + (size_t)col * K + k0 + kbB);
        acc = __builtin_amdgcn_wmma_f32_16x16x32_f16(false, a, false, b,
                                                     (short)0, acc, false, false);
    }
    const int rbase = mTile * 16 + (lane >> 4) * 8;   // C/D: VGPR r -> row rbase+r, col = lane&15
#pragma unroll
    for (int r = 0; r < 8; ++r)
        C[(size_t)(rbase + r) * N + col] = acc[r];
}

// ---------------- edge scatter: agg[tgt] += dinv[src] * h[src] ----------------
// one wave per edge; float4 vector loads, scalar f32 atomics (global_atomic_add_f32)
__global__ void k_scatter(const float* __restrict__ h, const float* __restrict__ dinv,
                          const int* __restrict__ src, const int* __restrict__ tgt,
                          float* __restrict__ agg, int nEdges, int F) {
    int e = blockIdx.x * (blockDim.x >> 5) + (threadIdx.x >> 5);
    int lane = threadIdx.x & 31;
    if (e >= nEdges) return;
    int s = src[e], t = tgt[e];
    float w = dinv[s];
    const float4* hs = (const float4*)(h + (size_t)s * F);
    float* at = agg + (size_t)t * F;
    for (int i = lane; i < (F >> 2); i += 32) {
        float4 v = hs[i];
        atomicAdd(at + 4 * i + 0, v.x * w);
        atomicAdd(at + 4 * i + 1, v.y * w);
        atomicAdd(at + 4 * i + 2, v.z * w);
        atomicAdd(at + 4 * i + 3, v.w * w);
    }
}

// ---------------- finalize: h = residual + BN(relu(dinv*agg + dinv^2*h + b)) (in place) --------
__global__ void k_finalize(float* __restrict__ h, const float* __restrict__ agg,
                           const float* __restrict__ dinv, const float* __restrict__ bias,
                           const float* __restrict__ g, const float* __restrict__ be,
                           const float* __restrict__ m, const float* __restrict__ v,
                           const float* __restrict__ residual, int nNodes, int F) {
    int i = blockIdx.x * blockDim.x + threadIdx.x;
    if (i >= nNodes * F) return;
    int n = i / F, f = i - n * F;
    float di = dinv[n];
    float y  = di * agg[i] + di * di * h[i] + bias[f];
    y = fmaxf(y, 0.0f);
    y = (y - m[f]) * (g[f] * rsqrtf(v[f] + BN_EPS)) + be[f];
    if (residual) y += residual[i];
    h[i] = y;
}

// ---------------- global mean pool ----------------
__global__ void k_pool(const float* __restrict__ x, const int* __restrict__ batch,
                       float* __restrict__ pooled, float* __restrict__ cnt, int nNodes, int F) {
    int n = blockIdx.x * (blockDim.x >> 5) + (threadIdx.x >> 5);
    int lane = threadIdx.x & 31;
    if (n >= nNodes) return;
    int gidx = batch[n];
    if (lane == 0) atomicAdd(&cnt[gidx], 1.0f);
    const float* xs = x + (size_t)n * F;
    float* pg = pooled + (size_t)gidx * F;
    for (int f = lane; f < F; f += 32) atomicAdd(&pg[f], xs[f]);
}
__global__ void k_pool_div(float* pooled, const float* __restrict__ cnt, int G, int F) {
    int i = blockIdx.x * blockDim.x + threadIdx.x;
    if (i >= G * F) return;
    pooled[i] /= fmaxf(cnt[i / F], 1.0f);
}

// ---------------- MLP head helpers ----------------
__global__ void k_bias_relu(float* h, const float* __restrict__ b, int total, int F) {
    int i = blockIdx.x * blockDim.x + threadIdx.x;
    if (i < total) h[i] = fmaxf(h[i] + b[i % F], 0.0f);
}
__global__ void k_matvec(const float* __restrict__ hm, const float* __restrict__ lw2,
                         const float* __restrict__ lb2, float* __restrict__ out, int G, int F) {
    int gidx = blockIdx.x * (blockDim.x >> 5) + (threadIdx.x >> 5);
    int lane = threadIdx.x & 31;
    if (gidx >= G) return;
    const float* row = hm + (size_t)gidx * F;
    float acc = 0.0f;
    for (int f = lane; f < F; f += 32) acc += row[f] * lw2[f];
#pragma unroll
    for (int off = 16; off > 0; off >>= 1) acc += __shfl_down(acc, off, 32);
    if (lane == 0) out[gidx] = acc + lb2[0];
}

extern "C" void kernel_launch(void* const* d_in, const int* in_sizes, int n_in,
                              void* d_out, int out_size, void* d_ws, size_t ws_size,
                              hipStream_t stream) {
    const float* x   = (const float*)d_in[0];
    const int* ei    = (const int*)d_in[1];
    const int* src   = ei;
    const int* tgt   = ei + N_EDGES;
    const int* batch = (const int*)d_in[2];
    const float* W1 = (const float*)d_in[3],  *b1 = (const float*)d_in[4];
    const float* W2 = (const float*)d_in[5],  *b2 = (const float*)d_in[6];
    const float* W3 = (const float*)d_in[7],  *b3 = (const float*)d_in[8];
    const float* g1 = (const float*)d_in[9],  *be1 = (const float*)d_in[10],
               * m1 = (const float*)d_in[11], *v1  = (const float*)d_in[12];
    const float* g2 = (const float*)d_in[13], *be2 = (const float*)d_in[14],
               * m2 = (const float*)d_in[15], *v2  = (const float*)d_in[16];
    const float* g3 = (const float*)d_in[17], *be3 = (const float*)d_in[18],
               * m3 = (const float*)d_in[19], *v3  = (const float*)d_in[20];
    const float* lw1 = (const float*)d_in[21], *lb1 = (const float*)d_in[22];
    const float* lw2 = (const float*)d_in[23], *lb2 = (const float*)d_in[24];
    float* out = (float*)d_out;

    // ---- workspace bump allocator (256B aligned) ----
    char* ws = (char*)d_ws;
    size_t off = 0;
    auto alloc = [&](size_t bytes) -> void* {
        void* p = ws + off;
        off += (bytes + 255) & ~(size_t)255;
        return p;
    };
    float*    dinv   = (float*)   alloc((size_t)N_NODES * 4);
    float*    buf0   = (float*)   alloc((size_t)N_NODES * HID * 4);
    float*    buf1   = (float*)   alloc((size_t)N_NODES * HID * 4);
    float*    agg    = (float*)   alloc((size_t)N_NODES * HID * 4);
    _Float16* x16    = (_Float16*)alloc((size_t)N_NODES * HID * 2);
    _Float16* w16    = (_Float16*)alloc((size_t)HID * HID * 2);
    float*    pooled = (float*)   alloc((size_t)N_GRAPHS * HID * 4);
    float*    cnt    = (float*)   alloc((size_t)N_GRAPHS * 4);
    float*    hmlp   = (float*)   alloc((size_t)N_GRAPHS * HID * 4);
    _Float16* p16    = (_Float16*)alloc((size_t)N_GRAPHS * HID * 2);
    (void)ws_size; (void)in_sizes; (void)n_in; (void)out_size;

    // ---- degree with self-loops -> dinv ----
    k_init_deg  <<<(N_NODES + 255) / 256, 256, 0, stream>>>(dinv, N_NODES);
    k_deg_accum <<<(N_EDGES + 255) / 256, 256, 0, stream>>>(tgt, dinv, N_EDGES);
    k_deg_to_dinv<<<(N_NODES + 255) / 256, 256, 0, stream>>>(dinv, N_NODES);

    auto gcn_layer = [&](const float* xin, int K, const float* W, const float* bias,
                         const float* g, const float* be, const float* m, const float* v,
                         const float* residual, float* hout) {
        int nx = N_NODES * K;
        k_f32_to_f16   <<<(nx + 255) / 256, 256, 0, stream>>>(xin, x16, nx);
        k_transpose_f16<<<(K * HID + 255) / 256, 256, 0, stream>>>(W, w16, K, HID);
        k_wmma_gemm    <<<N_NODES / 16, 512, 0, stream>>>(x16, w16, hout, N_NODES, HID, K);
        hipMemsetAsync(agg, 0, (size_t)N_NODES * HID * 4, stream);
        k_scatter      <<<(N_EDGES + 7) / 8, 256, 0, stream>>>(hout, dinv, src, tgt, agg,
                                                               N_EDGES, HID);
        k_finalize     <<<(N_NODES * HID + 255) / 256, 256, 0, stream>>>(
            hout, agg, dinv, bias, g, be, m, v, residual, N_NODES, HID);
    };

    gcn_layer(x,    IN_DIM, W1, b1, g1, be1, m1, v1, nullptr, buf0); // x1 -> buf0
    gcn_layer(buf0, HID,    W2, b2, g2, be2, m2, v2, buf0,    buf1); // x2 = x1 + ... -> buf1
    gcn_layer(buf1, HID,    W3, b3, g3, be3, m3, v3, buf1,    buf0); // x3 = x2 + ... -> buf0

    // ---- global mean pool ----
    hipMemsetAsync(pooled, 0, (size_t)N_GRAPHS * HID * 4, stream);
    hipMemsetAsync(cnt,    0, (size_t)N_GRAPHS * 4, stream);
    k_pool    <<<(N_NODES + 7) / 8, 256, 0, stream>>>(buf0, batch, pooled, cnt, N_NODES, HID);
    k_pool_div<<<(N_GRAPHS * HID + 255) / 256, 256, 0, stream>>>(pooled, cnt, N_GRAPHS, HID);

    // ---- MLP head ----
    k_f32_to_f16   <<<(N_GRAPHS * HID + 255) / 256, 256, 0, stream>>>(pooled, p16, N_GRAPHS * HID);
    k_transpose_f16<<<(HID * HID + 255) / 256, 256, 0, stream>>>(lw1, w16, HID, HID);
    k_wmma_gemm    <<<N_GRAPHS / 16, 512, 0, stream>>>(p16, w16, hmlp, N_GRAPHS, HID, HID);
    k_bias_relu    <<<(N_GRAPHS * HID + 255) / 256, 256, 0, stream>>>(hmlp, lb1,
                                                                      N_GRAPHS * HID, HID);
    k_matvec       <<<(N_GRAPHS * 32 + 255) / 256, 256, 0, stream>>>(hmlp, lw2, lb2, out,
                                                                     N_GRAPHS, HID);
}
